// MultiHeadSelfAttentionConv_64561948393705
// MI455X (gfx1250) — matrix operand order repaired and checked by reading, not compiled, not run on hardware
//
#include <hip/hip_runtime.h>
#include <hip/hip_bf16.h>

typedef __bf16 bf16_t;
typedef __attribute__((ext_vector_type(16))) __bf16 v16bf;
typedef __attribute__((ext_vector_type(8)))  __bf16 v8bf;
typedef __attribute__((ext_vector_type(8)))  float  v8f;
typedef unsigned int u32;
typedef unsigned long long u64;

#define BATCH 4
#define CDIM  512
#define LDIM  2048
#define HEADS 8
#define CH    64

// ---------- helpers ----------

__device__ __forceinline__ bf16_t f2bf(float f) {
  union { float f; u32 u; } v; v.f = f;
  u32 r = v.u + 0x7FFFu + ((v.u >> 16) & 1u);   // round-to-nearest-even
  unsigned short h = (unsigned short)(r >> 16);
  return __builtin_bit_cast(bf16_t, h);
}
__device__ __forceinline__ bf16_t bflo(u32 d) {
  return __builtin_bit_cast(bf16_t, (unsigned short)(d & 0xffffu));
}
__device__ __forceinline__ bf16_t bfhi(u32 d) {
  return __builtin_bit_cast(bf16_t, (unsigned short)(d >> 16));
}

__device__ __forceinline__ v8f wmma_bf16(v16bf a, v16bf b, v8f c) {
  return __builtin_amdgcn_wmma_f32_16x16x32_bf16(false, a, false, b,
                                                 (short)0, c, false, false);
}

// Async global->LDS copy of 16 bytes per lane (ASYNCcnt-tracked DMA path).
// LDS dest address = low 32 bits of the generic pointer (aperture is in [63:32]).
__device__ __forceinline__ void async_copy_b128(void* lds_dst, const void* gsrc) {
  u32 lds = (u32)(u64)lds_dst;
  asm volatile("global_load_async_to_lds_b128 %0, %1, off"
               :: "v"(lds), "v"(gsrc) : "memory");
}
__device__ __forceinline__ void wait_async0() {
  asm volatile("s_wait_asynccnt 0" ::: "memory");
}

// A-frag 16x32 (row-major [m][k] tile, pitch % 8 == 0):
// lane m=lane&15, hh=lane>>4; values = k in {hh*8..hh*8+7} ++ {16+hh*8..+7}
// -> two contiguous 16B loads (ds_load_b128 / global_load_b128).
__device__ __forceinline__ v16bf load_frag_a_w(const bf16_t* base, int pitch) {
  const int lane = threadIdx.x & 31;
  const int m = lane & 15, hh = lane >> 4;
  const bf16_t* p = base + m * pitch + hh * 8;
  v8bf lo = *(const v8bf*)(p);
  v8bf hi = *(const v8bf*)(p + 16);
  return __builtin_shufflevector(lo, hi, 0, 1, 2, 3, 4, 5, 6, 7,
                                 8, 9, 10, 11, 12, 13, 14, 15);
}

// B-frag 32x16 from a TRANSPOSED tile stored [n][k] row-major (pitch % 8 == 0):
// lane n=lane&15, hh=lane>>4; values = k in {hh*8..hh*8+7} ++ {16+hh*8..+7}.
__device__ __forceinline__ v16bf load_frag_b_w(const bf16_t* base, int pitch) {
  const int lane = threadIdx.x & 31;
  const int n = lane & 15, hh = lane >> 4;
  const bf16_t* p = base + n * pitch + hh * 8;
  v8bf lo = *(const v8bf*)(p);
  v8bf hi = *(const v8bf*)(p + 16);
  return __builtin_shufflevector(lo, hi, 0, 1, 2, 3, 4, 5, 6, 7,
                                 8, 9, 10, 11, 12, 13, 14, 15);
}

__device__ __forceinline__ float half_reduce_max(float v) {
#pragma unroll
  for (int off = 1; off < 16; off <<= 1) v = fmaxf(v, __shfl_xor(v, off, 32));
  return v;
}
__device__ __forceinline__ float half_reduce_sum(float v) {
#pragma unroll
  for (int off = 1; off < 16; off <<= 1) v += __shfl_xor(v, off, 32);
  return v;
}

// ---------- fp32 -> bf16 conversion ----------

__global__ void cvt_f32_bf16(const float* __restrict__ in,
                             bf16_t* __restrict__ out, int n) {
  for (int i = blockIdx.x * blockDim.x + threadIdx.x; i < n;
       i += gridDim.x * blockDim.x)
    out[i] = f2bf(in[i]);
}

// ---------- projection GEMM: out[o,l] = sum_c w[o,c]*x[b,c,l] + bias[o] ----------
// 128 threads = 4 waves; tile 64(M) x 64(N), K stepped by 32.

#define AP 40   // W tile pitch  (32 k + pad), multiple of 8
#define BP 40   // X^T tile pitch (32 k + pad), multiple of 8

__global__ __launch_bounds__(128)
void proj_gemm(const bf16_t* __restrict__ wb, const bf16_t* __restrict__ xb,
               const float* __restrict__ bias, bf16_t* __restrict__ outB,
               int qt_mode) {
  __shared__ __align__(16) bf16_t sA[64 * AP];  // W tile, [m][k]
  __shared__ __align__(16) bf16_t sB[64 * BP];  // X tile transposed, [n][k]

  const int l0 = blockIdx.x * 64;
  const int o0 = blockIdx.y * 64;
  const int b  = blockIdx.z;
  const uint4* w4 = (const uint4*)wb;                             // pitch CDIM/8
  const uint4* x4 = (const uint4*)(xb + (size_t)b * CDIM * LDIM); // pitch LDIM/8

  const int tid = threadIdx.x, wave = tid >> 5, lane = tid & 31;
  v8f acc[4] = {};

  for (int k0 = 0; k0 < CDIM; k0 += 32) {
    __syncthreads();
    // W tile: 64 rows x 4 uint4 -> async DMA copy, 2 x b128 per thread
#pragma unroll
    for (int t = 0; t < 2; ++t) {
      int idx = tid + t * 128;
      int r = idx >> 2, c4 = idx & 3;
      async_copy_b128(&sA[r * AP + c4 * 8],
                      w4 + (size_t)(o0 + r) * (CDIM / 8) + (k0 >> 3) + c4);
    }
    // X tile: batch the global loads, then transpose-store [n][k]
    uint4 d0, d1;
    {
      int idx = tid;
      d0 = x4[(size_t)(k0 + (idx >> 3)) * (LDIM / 8) + (l0 >> 3) + (idx & 7)];
      idx = tid + 128;
      d1 = x4[(size_t)(k0 + (idx >> 3)) * (LDIM / 8) + (l0 >> 3) + (idx & 7)];
    }
#pragma unroll
    for (int t = 0; t < 2; ++t) {
      uint4 d = t ? d1 : d0;
      int idx = tid + t * 128;
      int k = idx >> 3, n0 = (idx & 7) * 8;
      sB[(n0 + 0) * BP + k] = bflo(d.x); sB[(n0 + 1) * BP + k] = bfhi(d.x);
      sB[(n0 + 2) * BP + k] = bflo(d.y); sB[(n0 + 3) * BP + k] = bfhi(d.y);
      sB[(n0 + 4) * BP + k] = bflo(d.z); sB[(n0 + 5) * BP + k] = bfhi(d.z);
      sB[(n0 + 6) * BP + k] = bflo(d.w); sB[(n0 + 7) * BP + k] = bfhi(d.w);
    }
    wait_async0();
    __syncthreads();

    v16bf a = load_frag_a_w(&sA[wave * 16 * AP], AP);
#pragma unroll
    for (int j = 0; j < 4; ++j)
      acc[j] = wmma_bf16(a, load_frag_b_w(&sB[j * 16 * BP], BP), acc[j]);
  }

  const int hh = lane >> 4, n16 = lane & 15;
#pragma unroll
  for (int j = 0; j < 4; ++j) {
    int l = l0 + j * 16 + n16;
#pragma unroll
    for (int i = 0; i < 8; ++i) {
      int o = o0 + wave * 16 + i + 8 * hh;
      bf16_t hv = f2bf(acc[j][i] + bias[o]);
      if (qt_mode) {  // transposed store: (b, h, l, c) — used for Q AND K
        outB[(((size_t)b * HEADS + (o >> 6)) * LDIM + l) * CH + (o & 63)] = hv;
      } else {        // natural (b, o, l) — used for V (the .view quirk wants this)
        outB[((size_t)b * CDIM + o) * LDIM + l] = hv;
      }
    }
  }
}

// ---------- fused flash attention ----------
// 256 threads = 8 waves; wave owns 16 query rows of a 128-row block.
// KV streamed in 64-column tiles, double-buffered: async-DMA K copy and V
// global loads for tile i+1 are issued before computing tile i.

#define KP 72   // Kt tile pitch (64 + pad), multiple of 8
#define VP 72   // V^T tile pitch (64 m + pad), multiple of 8
#define PP 72   // P tile pitch (64 + pad), multiple of 8

__device__ __forceinline__ void stage_k_async(bf16_t* dstK, const bf16_t* src,
                                              int tid) {
  const uint4* s4 = (const uint4*)src;  // contiguous 8KB tile
#pragma unroll
  for (int t = 0; t < 2; ++t) {
    int idx = tid + t * 256;
    int r = idx >> 3, c4 = idx & 7;
    async_copy_b128(&dstK[r * KP + c4 * 8], s4 + idx);
  }
}

__device__ __forceinline__ void store_v_tr(bf16_t* dstV, uint4 d0, uint4 d1,
                                           int tid) {
#pragma unroll
  for (int t = 0; t < 2; ++t) {
    uint4 d = t ? d1 : d0;
    int idx = tid + t * 256;
    int m = idx >> 3, cc = (idx & 7) * 8;
    dstV[(cc + 0) * VP + m] = bflo(d.x); dstV[(cc + 1) * VP + m] = bfhi(d.x);
    dstV[(cc + 2) * VP + m] = bflo(d.y); dstV[(cc + 3) * VP + m] = bfhi(d.y);
    dstV[(cc + 4) * VP + m] = bflo(d.z); dstV[(cc + 5) * VP + m] = bfhi(d.z);
    dstV[(cc + 6) * VP + m] = bflo(d.w); dstV[(cc + 7) * VP + m] = bfhi(d.w);
  }
}

__global__ __launch_bounds__(256)
void attn_fused(const bf16_t* __restrict__ Qt, const bf16_t* __restrict__ Kt,
                const bf16_t* __restrict__ Vb, float* __restrict__ out) {
  __shared__ __align__(16) bf16_t sK[2][64 * KP];  // Kt tile [m][c], double-buffered
  __shared__ __align__(16) bf16_t sV[2][64 * VP];  // V2^T tile [cc][m], double-buffered
  __shared__ __align__(16) bf16_t sP[8][16 * PP];  // per-wave P tile [row][m]

  const int nblk = LDIM / 128;
  const int bh = blockIdx.x / nblk;                 // b*H + h
  const int l0 = (blockIdx.x % nblk) * 128;

  const bf16_t* qt = Qt + (size_t)bh * LDIM * CH;   // (l, c) row-major
  const bf16_t* kt = Kt + (size_t)bh * LDIM * CH;   // (m, c) row-major (K transposed)
  const bf16_t* vv = Vb + (size_t)bh * CH * LDIM;   // flat block == V2 (m, cc) row-major

  const int tid = threadIdx.x, wave = tid >> 5, lane = tid & 31;
  const int hh = lane >> 4, n16 = lane & 15;
  const int row0 = l0 + wave * 16;

  // loop-invariant Q fragments (global_load_b128 pairs)
  v16bf qa0 = load_frag_a_w(qt + (size_t)row0 * CH, CH);
  v16bf qa1 = load_frag_a_w(qt + (size_t)row0 * CH + 32, CH);

  v8f o0 = {}, o1 = {}, o2 = {}, o3 = {};
  float mrow[8], lrow[8];
#pragma unroll
  for (int i = 0; i < 8; ++i) { mrow[i] = -3.0e38f; lrow[i] = 0.0f; }

  bf16_t* sPw = sP[wave];
  const float scale = 0.125f;  // 1/sqrt(64)
  const int NIT = LDIM / 64;   // 32 KV tiles

  // prologue: stage tile 0 into buffer 0
  stage_k_async(sK[0], kt, tid);
  {
    const uint4* v4 = (const uint4*)vv;
    uint4 p0 = v4[tid], p1 = v4[tid + 256];
    store_v_tr(sV[0], p0, p1, tid);
  }

  for (int it = 0; it < NIT; ++it) {
    const int buf = it & 1;
    wait_async0();            // tile `it` K-DMA complete
    __syncthreads();          // tile `it` visible; prior readers of buf^1 done

    uint4 dv0, dv1;
    const bool havenext = (it + 1 < NIT);
    if (havenext) {           // issue tile it+1 into the other buffer
      stage_k_async(sK[buf ^ 1], kt + (size_t)(it + 1) * 64 * CH, tid);
      const uint4* v4n = (const uint4*)(vv + (size_t)(it + 1) * 64 * CH);
      dv0 = v4n[tid]; dv1 = v4n[tid + 256];
    }

    const bf16_t* cK = sK[buf];
    const bf16_t* cV = sV[buf];

    // S = Q^T K (16 x 64): 8 WMMAs
    v8f s0 = {}, s1 = {}, s2 = {}, s3 = {};
    s0 = wmma_bf16(qa0, load_frag_b_w(cK + 0 * 16 * KP, KP), s0);
    s0 = wmma_bf16(qa1, load_frag_b_w(cK + 0 * 16 * KP + 32, KP), s0);
    s1 = wmma_bf16(qa0, load_frag_b_w(cK + 1 * 16 * KP, KP), s1);
    s1 = wmma_bf16(qa1, load_frag_b_w(cK + 1 * 16 * KP + 32, KP), s1);
    s2 = wmma_bf16(qa0, load_frag_b_w(cK + 2 * 16 * KP, KP), s2);
    s2 = wmma_bf16(qa1, load_frag_b_w(cK + 2 * 16 * KP + 32, KP), s2);
    s3 = wmma_bf16(qa0, load_frag_b_w(cK + 3 * 16 * KP, KP), s3);
    s3 = wmma_bf16(qa1, load_frag_b_w(cK + 3 * 16 * KP + 32, KP), s3);

    // online softmax; C-frag row for VGPR i is (i + 8*hh)
#pragma unroll
    for (int i = 0; i < 8; ++i) {
      float a0 = s0[i] * scale, a1 = s1[i] * scale;
      float a2 = s2[i] * scale, a3 = s3[i] * scale;
      float rmax = half_reduce_max(fmaxf(fmaxf(a0, a1), fmaxf(a2, a3)));
      float mnew = fmaxf(mrow[i], rmax);
      float p0 = __expf(a0 - mnew), p1 = __expf(a1 - mnew);
      float p2 = __expf(a2 - mnew), p3 = __expf(a3 - mnew);
      float corr = __expf(mrow[i] - mnew);
      float rs = half_reduce_sum((p0 + p1) + (p2 + p3));
      lrow[i] = lrow[i] * corr + rs;
      mrow[i] = mnew;
      o0[i] *= corr; o1[i] *= corr; o2[i] *= corr; o3[i] *= corr;
      int row = i + 8 * hh;
      sPw[row * PP +  0 + n16] = f2bf(p0);
      sPw[row * PP + 16 + n16] = f2bf(p1);
      sPw[row * PP + 32 + n16] = f2bf(p2);
      sPw[row * PP + 48 + n16] = f2bf(p3);
    }
    // wave-private LDS region: drain DS counter for cross-lane visibility
    asm volatile("s_wait_dscnt 0" ::: "memory");

    // O += P @ V2 (16x64 * 64x64): 8 WMMAs
    v16bf pa0 = load_frag_a_w(sPw, PP);
    v16bf pa1 = load_frag_a_w(sPw + 32, PP);
    o0 = wmma_bf16(pa0, load_frag_b_w(cV + 0 * 16 * VP, VP), o0);
    o0 = wmma_bf16(pa1, load_frag_b_w(cV + 0 * 16 * VP + 32, VP), o0);
    o1 = wmma_bf16(pa0, load_frag_b_w(cV + 1 * 16 * VP, VP), o1);
    o1 = wmma_bf16(pa1, load_frag_b_w(cV + 1 * 16 * VP + 32, VP), o1);
    o2 = wmma_bf16(pa0, load_frag_b_w(cV + 2 * 16 * VP, VP), o2);
    o2 = wmma_bf16(pa1, load_frag_b_w(cV + 2 * 16 * VP + 32, VP), o2);
    o3 = wmma_bf16(pa0, load_frag_b_w(cV + 3 * 16 * VP, VP), o3);
    o3 = wmma_bf16(pa1, load_frag_b_w(cV + 3 * 16 * VP + 32, VP), o3);

    if (havenext)  // transpose-store V tile it+1 into the other buffer
      store_v_tr(sV[buf ^ 1], dv0, dv1, tid);
  }

  // epilogue: normalize and store; out.reshape(B,C,L) quirk == flat (bh, l, c)
#pragma unroll
  for (int i = 0; i < 8; ++i) {
    float inv = 1.0f / lrow[i];
    size_t l = (size_t)(row0 + i + 8 * hh);
    float* op = out + ((size_t)bh * LDIM + l) * CH;
    op[ 0 + n16] = o0[i] * inv;
    op[16 + n16] = o1[i] * inv;
    op[32 + n16] = o2[i] * inv;
    op[48 + n16] = o3[i] * inv;
  }
}

// ---------- launch ----------

extern "C" void kernel_launch(void* const* d_in, const int* in_sizes, int n_in,
                              void* d_out, int out_size, void* d_ws, size_t ws_size,
                              hipStream_t stream) {
  const float* x  = (const float*)d_in[0];
  const float* wq = (const float*)d_in[1];
  const float* bq = (const float*)d_in[2];
  const float* wk = (const float*)d_in[3];
  const float* bk = (const float*)d_in[4];
  const float* wv = (const float*)d_in[5];
  const float* bv = (const float*)d_in[6];
  float* out = (float*)d_out;

  const size_t NX = (size_t)BATCH * CDIM * LDIM;  // 4,194,304
  const size_t NW = (size_t)CDIM * CDIM;          // 262,144

  bf16_t* p = (bf16_t*)d_ws;
  bf16_t* xb  = p; p += NX;
  bf16_t* wqb = p; p += NW;
  bf16_t* wkb = p; p += NW;
  bf16_t* wvb = p; p += NW;
  bf16_t* Qt  = p; p += NX;   // (b, h, l, c)
  bf16_t* Kt  = p; p += NX;   // (b, h, l, c)  (K transposed)
  bf16_t* Vb  = p; p += NX;   // (b, c, l) == per-head V2 (m, cc)

  cvt_f32_bf16<<<4096, 256, 0, stream>>>(x,  xb,  (int)NX);
  cvt_f32_bf16<<<512,  256, 0, stream>>>(wq, wqb, (int)NW);
  cvt_f32_bf16<<<512,  256, 0, stream>>>(wk, wkb, (int)NW);
  cvt_f32_bf16<<<512,  256, 0, stream>>>(wv, wvb, (int)NW);

  dim3 pg(LDIM / 64, CDIM / 64, BATCH);
  proj_gemm<<<pg, 128, 0, stream>>>(wqb, xb, bq, Qt, 1);
  proj_gemm<<<pg, 128, 0, stream>>>(wkb, xb, bk, Kt, 1);
  proj_gemm<<<pg, 128, 0, stream>>>(wvb, xb, bv, Vb, 0);

  attn_fused<<<BATCH * HEADS * (LDIM / 128), 256, 0, stream>>>(Qt, Kt, Vb, out);
}
